// FasterRCNN_42683384987715
// MI455X (gfx1250) — matrix operand order repaired and checked by reading, not compiled, not run on hardware
//
#include <hip/hip_runtime.h>
#include <math.h>

// ---------------------------------------------------------------------------
// Types
// ---------------------------------------------------------------------------
typedef __bf16 bf16;
typedef __attribute__((ext_vector_type(16))) __bf16 v16bf;
typedef __attribute__((ext_vector_type(8)))  float  v8f;

__device__ __forceinline__ bf16 f2b(float f) {
    unsigned u = __builtin_bit_cast(unsigned, f);
    unsigned r = (u + 0x7FFFu + ((u >> 16) & 1u)) >> 16;   // RNE
    unsigned short s = (unsigned short)r;
    return __builtin_bit_cast(bf16, s);
}
__device__ __forceinline__ float b2f(bf16 b) {
    unsigned short s = __builtin_bit_cast(unsigned short, b);
    unsigned u = ((unsigned)s) << 16;
    return __builtin_bit_cast(float, u);
}

#define IMG_H   600
#define IMG_W   800
#define BATCH   2
#define FH      37
#define FW      50
#define NANCH   9
#define TOTAL_A (FH * FW * NANCH)     // 16650
#define NPAD    32768
#define PRE_NMS 6000
#define POST_NMS 300
#define NMS_TH  0.7f
#define MIN_SZ  16.0f
#define ROI_SZ  7
#define POOLK   (512 * ROI_SZ * ROI_SZ)  // 25088
#define NEG_INF (-__builtin_inff())

// d_out layout (floats): cls_loc[600*84] | scores[600*21] | rois[600*4] | idx[600]
#define OUT_CLS   0
#define OUT_SC    (600 * 84)           // 50400
#define OUT_ROIS  (OUT_SC + 600 * 21)  // 63000
#define OUT_IDX   (OUT_ROIS + 600 * 4) // 65400

#define WMMA_BF16(A, B, C) \
    __builtin_amdgcn_wmma_f32_16x16x32_bf16(false, (A), false, (B), (short)0, (C), false, false)

// ---------------------------------------------------------------------------
// fp32 -> bf16 conversion
// ---------------------------------------------------------------------------
__global__ void k_f2bf(const float* __restrict__ s, bf16* __restrict__ d, size_t n) {
    size_t i = (size_t)blockIdx.x * blockDim.x + threadIdx.x;
    size_t st = (size_t)gridDim.x * blockDim.x;
    for (; i < n; i += st) d[i] = f2b(s[i]);
}

// ---------------------------------------------------------------------------
// 2x2 maxpool stride 2 (floor), NCHW bf16
// ---------------------------------------------------------------------------
__global__ void k_maxpool2(const bf16* __restrict__ in, bf16* __restrict__ out,
                           int C, int H, int W) {
    int Ho = H / 2, Wo = W / 2;
    size_t n = (size_t)BATCH * C * Ho * Wo;
    size_t i = (size_t)blockIdx.x * blockDim.x + threadIdx.x;
    size_t st = (size_t)gridDim.x * blockDim.x;
    for (; i < n; i += st) {
        int wo = (int)(i % Wo); size_t t = i / Wo;
        int ho = (int)(t % Ho); t /= Ho;
        int c = (int)(t % C); int b = (int)(t / C);
        const bf16* p = in + ((size_t)(b * C + c) * H + (size_t)ho * 2) * W + (size_t)wo * 2;
        float m = fmaxf(fmaxf(b2f(p[0]), b2f(p[1])), fmaxf(b2f(p[W]), b2f(p[W + 1])));
        out[i] = f2b(m);
    }
}

// ---------------------------------------------------------------------------
// Implicit-GEMM conv (SAME, stride 1, KH in {1,3}), bf16 WMMA.
// Block: 256 threads = 8 waves arranged 2(M) x 4(N).
// Block tile 64 (Cout) x 256 (pixels); wave tile 32x64 = 2x4 WMMA tiles
// (8 v_wmma per K-chunk per wave). LDS holds tiles in FRAGMENT-MAJOR order:
// each lane's 16-element fragment is one contiguous 32B run -> 2x ds_load_b128.
//   A(16x32): (r,k) -> lane (k>>4)*16 + (r&15),    elem  k&15
//   B(32x16): (k,n) -> lane (k>>4)*16 + (n&15),    elem  k&15
// ---------------------------------------------------------------------------
__global__ void k_conv_wmma(const bf16* __restrict__ in, const bf16* __restrict__ wgt,
                            const float* __restrict__ bias, void* __restrict__ out,
                            int Cin, int H, int W, int Cout, int KH,
                            int relu, int outF32) {
    const int HW = H * W;
    const int KK = KH * KH;
    const int K = Cin * KK;
    const int pad = (KH == 3) ? 1 : 0;
    const int b = blockIdx.z;
    const int px0 = blockIdx.x * 256;
    const int co0 = blockIdx.y * 64;

    __shared__ __align__(16) bf16 As2[4][32][16];    // 4 M-tiles
    __shared__ __align__(16) bf16 Bs2[16][32][16];   // 16 N-tiles
    __shared__ short sPy[256], sPx[256];

    const int tid = threadIdx.x;
    const int wid = tid >> 5, lane = tid & 31;
    const int hi = lane >> 4, mr = lane & 15;
    const int wm = (wid >> 2) * 32;   // 0,32
    const int wn = (wid & 3) * 64;    // 0,64,128,192

    // per-block pixel coordinate decode (once)
    {
        int px = px0 + tid;
        int y = px / W;
        sPy[tid] = (short)y;
        sPx[tid] = (short)(px - y * W);
    }
    __syncthreads();

    v8f acc[2][4] = {};

    for (int kc = 0; kc < K; kc += 32) {
        // ---- stage A: weights [Cout x K], 64x32 chunk, fragment-major ----
#pragma unroll
        for (int it = 0; it < 8; it++) {
            int r = (tid >> 5) + it * 8;
            int kk = tid & 31;
            int gco = co0 + r, gk = kc + kk;
            bf16 v = (gco < Cout && gk < K) ? wgt[(size_t)gco * K + gk] : f2b(0.f);
            As2[r >> 4][(kk >> 4) * 16 + (r & 15)][kk & 15] = v;
        }
        // ---- stage B: im2col 32(k) x 256(px), one k per thread ----
        {
            int kk = tid >> 3;                 // 0..31
            int gk = kc + kk;
            bool kvalid = gk < K;
            int cBase = 0, dy = 0, dx = 0;
            if (kvalid) {
                int ci = gk / KK;
                int rem = gk - ci * KK;
                int kh = rem / KH;
                dy = kh - pad;
                dx = (rem - kh * KH) - pad;
                cBase = (b * Cin + ci) * HW;
            }
            int laneHalf = (kk >> 4) * 16, eB = kk & 15;
#pragma unroll
            for (int it = 0; it < 32; it++) {
                int nn = (tid & 7) + it * 8;
                bf16 v = f2b(0.f);
                int px = px0 + nn;
                if (kvalid && px < HW) {
                    int yy = sPy[nn] + dy, xx = sPx[nn] + dx;
                    if (yy >= 0 && yy < H && xx >= 0 && xx < W)
                        v = in[(size_t)cBase + (size_t)yy * W + xx];
                }
                Bs2[nn >> 4][laneHalf + (nn & 15)][eB] = v;
            }
        }
        if (kc + 32 < K)
            __builtin_prefetch(wgt + (size_t)(co0 + (tid & 63)) * K + (kc + 32), 0, 1);
        __syncthreads();

        // ---- fragments: one contiguous 32B vector load per lane per tile ----
        v16bf a0 = *(const v16bf*)&As2[(wm >> 4) + 0][lane][0];
        v16bf a1 = *(const v16bf*)&As2[(wm >> 4) + 1][lane][0];
#pragma unroll
        for (int tj = 0; tj < 4; tj++) {
            v16bf bf_ = *(const v16bf*)&Bs2[(wn >> 4) + tj][lane][0];
            acc[0][tj] = WMMA_BF16(a0, bf_, acc[0][tj]);
            acc[1][tj] = WMMA_BF16(a1, bf_, acc[1][tj]);
        }
        __syncthreads();
    }

#pragma unroll
    for (int ti = 0; ti < 2; ti++)
#pragma unroll
        for (int tj = 0; tj < 4; tj++)
#pragma unroll
            for (int r = 0; r < 8; r++) {
                int ch = co0 + wm + ti * 16 + r + hi * 8;
                int px = px0 + wn + tj * 16 + mr;
                if (ch < Cout && px < HW) {
                    float v = acc[ti][tj][r] + bias[ch];
                    if (relu) v = fmaxf(v, 0.f);
                    size_t o = ((size_t)(b * Cout + ch)) * HW + px;
                    if (outF32) ((float*)out)[o] = v;
                    else        ((bf16*)out)[o] = f2b(v);
                }
            }
}

// ---------------------------------------------------------------------------
// Generic GEMM: D[MxN] = A[MxK] * B[KxN] + bias[N], bf16 in, WMMA bf16.
// Same 64x256 block tile / 32x64 wave tile / fragment-major LDS as the conv.
// ---------------------------------------------------------------------------
__global__ void k_gemm_wmma(const bf16* __restrict__ A, const bf16* __restrict__ B,
                            const float* __restrict__ bias, void* __restrict__ out,
                            int M, int N, int K, int relu, int outF32, int ldo) {
    __shared__ __align__(16) bf16 As2[4][32][16];
    __shared__ __align__(16) bf16 Bs2[16][32][16];
    const int tid = threadIdx.x;
    const int wid = tid >> 5, lane = tid & 31;
    const int hi = lane >> 4, mr = lane & 15;
    const int m0 = blockIdx.y * 64, n0 = blockIdx.x * 256;
    const int wm = (wid >> 2) * 32;
    const int wn = (wid & 3) * 64;

    v8f acc[2][4] = {};
    for (int kc = 0; kc < K; kc += 32) {
#pragma unroll
        for (int it = 0; it < 8; it++) {
            int r = (tid >> 5) + it * 8;
            int kk = tid & 31;
            int gr = m0 + r, gk = kc + kk;
            bf16 v = (gr < M && gk < K) ? A[(size_t)gr * K + gk] : f2b(0.f);
            As2[r >> 4][(kk >> 4) * 16 + (r & 15)][kk & 15] = v;
        }
#pragma unroll
        for (int it = 0; it < 32; it++) {
            int t = it * 256 + tid;
            int kk = t >> 8, nn = t & 255;
            int gk = kc + kk, gn = n0 + nn;
            bf16 v = (gk < K && gn < N) ? B[(size_t)gk * N + gn] : f2b(0.f);
            Bs2[nn >> 4][(kk >> 4) * 16 + (nn & 15)][kk & 15] = v;
        }
        if (kc + 32 < K) {
            __builtin_prefetch(A + (size_t)(m0 + (tid & 63)) * K + (kc + 32), 0, 1);
            __builtin_prefetch(B + (size_t)(kc + 32 + (tid >> 3)) * N + n0 + (tid & 7) * 32, 0, 1);
        }
        __syncthreads();

        v16bf a0 = *(const v16bf*)&As2[(wm >> 4) + 0][lane][0];
        v16bf a1 = *(const v16bf*)&As2[(wm >> 4) + 1][lane][0];
#pragma unroll
        for (int tj = 0; tj < 4; tj++) {
            v16bf bf_ = *(const v16bf*)&Bs2[(wn >> 4) + tj][lane][0];
            acc[0][tj] = WMMA_BF16(a0, bf_, acc[0][tj]);
            acc[1][tj] = WMMA_BF16(a1, bf_, acc[1][tj]);
        }
        __syncthreads();
    }
#pragma unroll
    for (int ti = 0; ti < 2; ti++)
#pragma unroll
        for (int tj = 0; tj < 4; tj++)
#pragma unroll
            for (int r = 0; r < 8; r++) {
                int row = m0 + wm + ti * 16 + r + hi * 8;
                int col = n0 + wn + tj * 16 + mr;
                if (row < M && col < N) {
                    float v = acc[ti][tj][r] + bias[col];
                    if (relu) v = fmaxf(v, 0.f);
                    if (outF32) ((float*)out)[(size_t)row * ldo + col] = v;
                    else        ((bf16*)out)[(size_t)row * ldo + col] = f2b(v);
                }
            }
}

// ---------------------------------------------------------------------------
// Anchor decode + clip + min-size mask -> boxes + sort keys (padded to NPAD)
// ---------------------------------------------------------------------------
__global__ void k_boxes(const float* __restrict__ locmap, const float* __restrict__ scmap,
                        float* __restrict__ boxes, float* __restrict__ keys,
                        int* __restrict__ vals) {
    int b = blockIdx.y;
    int a = blockIdx.x * blockDim.x + threadIdx.x;
    if (a >= NPAD) return;
    float* kb = keys + (size_t)b * NPAD;
    int*   vb = vals + (size_t)b * NPAD;
    if (a >= TOTAL_A) { kb[a] = NEG_INF; vb[a] = 0; return; }

    int cell = a / NANCH, t = a % NANCH;
    int y = cell / FW, x = cell % FW;
    int ri = t / 3, si = t % 3;
    const float ratios[3] = {0.5f, 1.0f, 2.0f};
    const float scales[3] = {8.0f, 16.0f, 32.0f};
    float ah = 16.0f * scales[si] * sqrtf(ratios[ri]);
    float aw = 16.0f * scales[si] * sqrtf(1.0f / ratios[ri]);
    float acy = (float)(y * 16) + 8.0f;
    float acx = (float)(x * 16) + 8.0f;

    const int HW = FH * FW;
    const float* L = locmap + (size_t)b * 36 * HW;
    size_t pix = (size_t)y * FW + x;
    float dy = L[(size_t)(t * 4 + 0) * HW + pix];
    float dx = L[(size_t)(t * 4 + 1) * HW + pix];
    float dh = L[(size_t)(t * 4 + 2) * HW + pix];
    float dw = L[(size_t)(t * 4 + 3) * HW + pix];

    float cy = dy * ah + acy, cx = dx * aw + acx;
    float hh = expf(dh) * ah, ww = expf(dw) * aw;
    float y1 = fminf(fmaxf(cy - 0.5f * hh, 0.f), (float)IMG_H);
    float x1 = fminf(fmaxf(cx - 0.5f * ww, 0.f), (float)IMG_W);
    float y2 = fminf(fmaxf(cy + 0.5f * hh, 0.f), (float)IMG_H);
    float x2 = fminf(fmaxf(cx + 0.5f * ww, 0.f), (float)IMG_W);

    bool ok = (y2 - y1 >= MIN_SZ) && (x2 - x1 >= MIN_SZ);
    float fg = scmap[((size_t)b * 18 + (t * 2 + 1)) * HW + pix];
    float* bx = boxes + ((size_t)b * TOTAL_A + a) * 4;
    bx[0] = y1; bx[1] = x1; bx[2] = y2; bx[3] = x2;
    kb[a] = ok ? fg : NEG_INF;
    vb[a] = a;
}

// ---------------------------------------------------------------------------
// Global-memory bitonic sort step (descending by key), per batch (blockIdx.y)
// ---------------------------------------------------------------------------
__global__ void k_bitonic(float* __restrict__ keys, int* __restrict__ vals,
                          int n, int k, int j) {
    int b = blockIdx.y;
    int i = blockIdx.x * blockDim.x + threadIdx.x;
    if (i >= n) return;
    float* kb = keys + (size_t)b * n;
    int*   vb = vals + (size_t)b * n;
    int ixj = i ^ j;
    if (ixj > i) {
        float ki = kb[i], kj = kb[ixj];
        bool descSeg = ((i & k) == 0);
        bool doswap = descSeg ? (ki < kj) : (ki > kj);
        if (doswap) {
            kb[i] = kj; kb[ixj] = ki;
            int vi = vb[i]; vb[i] = vb[ixj]; vb[ixj] = vi;
        }
    }
}

// ---------------------------------------------------------------------------
// Gather top PRE_NMS sorted boxes + keys
// ---------------------------------------------------------------------------
__global__ void k_gather(const float* __restrict__ keys, const int* __restrict__ vals,
                         const float* __restrict__ boxes,
                         float* __restrict__ tb, float* __restrict__ tk) {
    int b = blockIdx.y;
    int i = blockIdx.x * blockDim.x + threadIdx.x;
    if (i >= PRE_NMS) return;
    int v = vals[(size_t)b * NPAD + i];
    const float* bx = boxes + ((size_t)b * TOTAL_A + v) * 4;
    float* o = tb + ((size_t)b * PRE_NMS + i) * 4;
    o[0] = bx[0]; o[1] = bx[1]; o[2] = bx[2]; o[3] = bx[3];
    tk[(size_t)b * PRE_NMS + i] = keys[(size_t)b * NPAD + i];
}

// ---------------------------------------------------------------------------
// Serial-outer / parallel-inner NMS; one block per batch (blockIdx.y)
// ---------------------------------------------------------------------------
__global__ void k_nms(const float* __restrict__ tb, int* __restrict__ keep) {
    __shared__ int keepS[PRE_NMS];
    int b = blockIdx.y;
    const float* T = tb + (size_t)b * PRE_NMS * 4;
    for (int i = threadIdx.x; i < PRE_NMS; i += blockDim.x) keepS[i] = 1;
    __syncthreads();
    for (int i = 0; i < PRE_NMS; i++) {
        if (keepS[i]) {
            float y1 = T[i * 4 + 0], x1 = T[i * 4 + 1];
            float y2 = T[i * 4 + 2], x2 = T[i * 4 + 3];
            float ai = (y2 - y1) * (x2 - x1);
            for (int j = i + 1 + threadIdx.x; j < PRE_NMS; j += blockDim.x) {
                float yy1 = fmaxf(y1, T[j * 4 + 0]);
                float xx1 = fmaxf(x1, T[j * 4 + 1]);
                float yy2 = fminf(y2, T[j * 4 + 2]);
                float xx2 = fminf(x2, T[j * 4 + 3]);
                float inter = fmaxf(yy2 - yy1, 0.f) * fmaxf(xx2 - xx1, 0.f);
                float aj = (T[j * 4 + 2] - T[j * 4 + 0]) * (T[j * 4 + 3] - T[j * 4 + 1]);
                float iou = inter / fmaxf(ai + aj - inter, 1e-9f);
                if (iou > NMS_TH) keepS[j] = 0;
            }
        }
        __syncthreads();
    }
    for (int i = threadIdx.x; i < PRE_NMS; i += blockDim.x)
        keep[(size_t)b * PRE_NMS + i] = keepS[i];
}

// ---------------------------------------------------------------------------
// Ordered selection of POST_NMS kept+finite rois -> rois scratch + d_out
// ---------------------------------------------------------------------------
__global__ void k_select(const float* __restrict__ tb, const float* __restrict__ tk,
                         const int* __restrict__ keep, float* __restrict__ rois,
                         float* __restrict__ out) {
    if (threadIdx.x != 0 || blockIdx.x != 0) return;
    for (int b = 0; b < BATCH; b++) {
        const float* T = tb + (size_t)b * PRE_NMS * 4;
        const float* K = tk + (size_t)b * PRE_NMS;
        const int* KP = keep + (size_t)b * PRE_NMS;
        int cnt = 0;
        for (int i = 0; i < PRE_NMS && cnt < POST_NMS; i++) {
            float s = K[i];
            if (KP[i] && (s == s) && s > -1e37f && s < 1e37f) {
                float* R = rois + ((size_t)b * POST_NMS + cnt) * 4;
                R[0] = T[i * 4 + 0]; R[1] = T[i * 4 + 1];
                R[2] = T[i * 4 + 2]; R[3] = T[i * 4 + 3];
                cnt++;
            }
        }
        for (int j = cnt; j < POST_NMS; j++) {
            float* R = rois + ((size_t)b * POST_NMS + j) * 4;
            R[0] = R[1] = R[2] = R[3] = 0.f;
        }
        for (int j = 0; j < POST_NMS; j++) {
            const float* R = rois + ((size_t)b * POST_NMS + j) * 4;
            float* O = out + OUT_ROIS + ((size_t)b * POST_NMS + j) * 4;
            O[0] = R[0]; O[1] = R[1]; O[2] = R[2]; O[3] = R[3];
            out[OUT_IDX + b * POST_NMS + j] = (float)b;
        }
    }
}

// ---------------------------------------------------------------------------
// ROI max-pool 7x7 with 2x2 samples per bin. One block per roi.
// ---------------------------------------------------------------------------
__global__ void k_roipool(const bf16* __restrict__ feat, const float* __restrict__ rois,
                          bf16* __restrict__ pool) {
    int r = blockIdx.x;
    int b = r / POST_NMS;
    __shared__ int sIdx[2][ROI_SZ][2];   // [y/x][bin][sample]
    if (threadIdx.x < 28) {
        int t = threadIdx.x;
        int dim = t / 14;               // 0 = y, 1 = x
        int rem = t % 14;
        int bin = rem / 2, s = rem % 2;
        float off = s ? 0.75f : 0.25f;
        float c1 = rois[(size_t)r * 4 + dim] / 16.0f;
        float c2 = rois[(size_t)r * 4 + 2 + dim] / 16.0f;
        float v = c1 + ((float)bin + off) * ((c2 - c1) / (float)ROI_SZ);
        int lim = dim ? (FW - 1) : (FH - 1);
        int iv = (int)floorf(v);
        iv = iv < 0 ? 0 : (iv > lim ? lim : iv);
        sIdx[dim][bin][s] = iv;
    }
    __syncthreads();
    for (int idx = threadIdx.x; idx < POOLK; idx += blockDim.x) {
        int c = idx / (ROI_SZ * ROI_SZ);
        int bin = idx % (ROI_SZ * ROI_SZ);
        int by = bin / ROI_SZ, bx = bin % ROI_SZ;
        const bf16* F = feat + ((size_t)(b * 512 + c)) * (FH * FW);
        float m = NEG_INF;
#pragma unroll
        for (int sy = 0; sy < 2; sy++)
#pragma unroll
            for (int sx = 0; sx < 2; sx++) {
                float v = b2f(F[(size_t)sIdx[0][by][sy] * FW + sIdx[1][bx][sx]]);
                m = fmaxf(m, v);
            }
        pool[(size_t)r * POOLK + idx] = f2b(m);
    }
}

// ---------------------------------------------------------------------------
// Host side
// ---------------------------------------------------------------------------
static inline unsigned cdiv(unsigned a, unsigned b) { return (a + b - 1) / b; }

extern "C" void kernel_launch(void* const* d_in, const int* in_sizes, int n_in,
                              void* d_out, int out_size, void* d_ws, size_t ws_size,
                              hipStream_t stream) {
    (void)in_sizes; (void)n_in; (void)out_size; (void)ws_size;

    // ---- inputs (setup_inputs dict insertion order, tuples flattened) ----
    const float* X = (const float*)d_in[0];
    const float* convW[13]; const float* convB[13];
    for (int i = 0; i < 13; i++) { convW[i] = (const float*)d_in[1 + i]; convB[i] = (const float*)d_in[14 + i]; }
    const float* rpnW  = (const float*)d_in[27]; const float* rpnB  = (const float*)d_in[28];
    const float* scW   = (const float*)d_in[29]; const float* scB   = (const float*)d_in[30];
    const float* locW  = (const float*)d_in[31]; const float* locB  = (const float*)d_in[32];
    const float* fc1W  = (const float*)d_in[33]; const float* fc1B  = (const float*)d_in[34];
    const float* fc2W  = (const float*)d_in[35]; const float* fc2B  = (const float*)d_in[36];
    const float* clW   = (const float*)d_in[37]; const float* clB   = (const float*)d_in[38];
    const float* sfW   = (const float*)d_in[39]; const float* sfB   = (const float*)d_in[40];

    // ---- workspace layout ----
    const size_t MBYTE = (size_t)1 << 20;
    char* ws = (char*)d_ws;
    bf16*  A0    = (bf16*)(ws + 0);              // 128 MB activation ping
    bf16*  A1    = (bf16*)(ws + 128 * MBYTE);    // 128 MB activation pong
    bf16*  WB    = (bf16*)(ws + 256 * MBYTE);    // 200 MB bf16 weight staging
    float* LOC   = (float*)(ws + 456 * MBYTE);   // [B,36,FH,FW]
    float* SC    = (float*)(ws + 457 * MBYTE);   // [B,18,FH,FW]
    float* BOX   = (float*)(ws + 458 * MBYTE);   // [B,TOTAL_A,4]
    float* KEY   = (float*)(ws + 459 * MBYTE);   // [B,NPAD]
    int*   VAL   = (int*)  (ws + 460 * MBYTE);   // [B,NPAD]
    float* TB    = (float*)(ws + 461 * MBYTE);   // [B,PRE_NMS,4]
    float* TK    = (float*)(ws + 462 * MBYTE);   // [B,PRE_NMS]
    int*   KEEP  = (int*)  (ws + 463 * MBYTE);   // [B,PRE_NMS]
    float* ROIS  = (float*)(ws + 464 * MBYTE);   // [600,4]
    bf16*  POOL  = (bf16*) (ws + 465 * MBYTE);   // [600,25088]
    bf16*  FC1   = (bf16*) (ws + 497 * MBYTE);   // [600,4096]
    bf16*  FC2   = (bf16*) (ws + 505 * MBYTE);   // [600,4096]
    float* OUT   = (float*)d_out;

    // ---- backbone ----
    {
        size_t n = (size_t)BATCH * 3 * IMG_H * IMG_W;
        k_f2bf<<<cdiv((unsigned)n, 256), 256, 0, stream>>>(X, A0, n);
    }
    static const int CFG[17] = {64, 64, -1, 128, 128, -1, 256, 256, 256, -1,
                                512, 512, 512, -1, 512, 512, 512};
    bf16* bufs[2] = {A0, A1};
    int p = 0, Cin = 3, H = IMG_H, W = IMG_W, ci = 0;
    for (int li = 0; li < 17; li++) {
        if (CFG[li] < 0) {
            size_t n = (size_t)BATCH * Cin * (H / 2) * (W / 2);
            k_maxpool2<<<cdiv((unsigned)n, 256), 256, 0, stream>>>(bufs[p], bufs[p ^ 1], Cin, H, W);
            H /= 2; W /= 2; p ^= 1;
        } else {
            int Cout = CFG[li];
            size_t wn = (size_t)Cout * Cin * 9;
            k_f2bf<<<cdiv((unsigned)wn, 256), 256, 0, stream>>>(convW[ci], WB, wn);
            dim3 g(cdiv((unsigned)(H * W), 256), cdiv((unsigned)Cout, 64), BATCH);
            k_conv_wmma<<<g, 256, 0, stream>>>(bufs[p], WB, convB[ci], bufs[p ^ 1],
                                               Cin, H, W, Cout, 3, 1, 0);
            Cin = Cout; p ^= 1; ci++;
        }
    }
    bf16* FEAT = bufs[p];        // [B,512,FH,FW]
    bf16* RPN  = bufs[p ^ 1];

    // ---- RPN head ----
    {
        size_t wn = (size_t)512 * 512 * 9;
        k_f2bf<<<cdiv((unsigned)wn, 256), 256, 0, stream>>>(rpnW, WB, wn);
        dim3 g(cdiv(FH * FW, 256), cdiv(512u, 64), BATCH);
        k_conv_wmma<<<g, 256, 0, stream>>>(FEAT, WB, rpnB, RPN, 512, FH, FW, 512, 3, 1, 0);
    }
    {
        size_t wn = (size_t)36 * 512;
        k_f2bf<<<cdiv((unsigned)wn, 256), 256, 0, stream>>>(locW, WB, wn);
        dim3 g(cdiv(FH * FW, 256), 1, BATCH);
        k_conv_wmma<<<g, 256, 0, stream>>>(RPN, WB, locB, LOC, 512, FH, FW, 36, 1, 0, 1);
    }
    {
        size_t wn = (size_t)18 * 512;
        k_f2bf<<<cdiv((unsigned)wn, 256), 256, 0, stream>>>(scW, WB, wn);
        dim3 g(cdiv(FH * FW, 256), 1, BATCH);
        k_conv_wmma<<<g, 256, 0, stream>>>(RPN, WB, scB, SC, 512, FH, FW, 18, 1, 0, 1);
    }

    // ---- proposals ----
    k_boxes<<<dim3(cdiv(NPAD, 256), BATCH), 256, 0, stream>>>(LOC, SC, BOX, KEY, VAL);
    for (int k = 2; k <= NPAD; k <<= 1)
        for (int j = k >> 1; j > 0; j >>= 1)
            k_bitonic<<<dim3(NPAD / 256, BATCH), 256, 0, stream>>>(KEY, VAL, NPAD, k, j);
    k_gather<<<dim3(cdiv(PRE_NMS, 256), BATCH), 256, 0, stream>>>(KEY, VAL, BOX, TB, TK);
    k_nms<<<dim3(1, BATCH), 256, 0, stream>>>(TB, KEEP);
    k_select<<<1, 1, 0, stream>>>(TB, TK, KEEP, ROIS, OUT);

    // ---- ROI pool ----
    k_roipool<<<BATCH * POST_NMS, 256, 0, stream>>>(FEAT, ROIS, POOL);

    // ---- FC head (WMMA GEMMs) ----
    const int M = BATCH * POST_NMS;   // 600
    {
        size_t wn = (size_t)POOLK * 4096;
        k_f2bf<<<cdiv((unsigned)wn, 256), 256, 0, stream>>>(fc1W, WB, wn);
        dim3 g(cdiv(4096u, 256), cdiv((unsigned)M, 64));
        k_gemm_wmma<<<g, 256, 0, stream>>>(POOL, WB, fc1B, FC1, M, 4096, POOLK, 1, 0, 4096);
    }
    {
        size_t wn = (size_t)4096 * 4096;
        k_f2bf<<<cdiv((unsigned)wn, 256), 256, 0, stream>>>(fc2W, WB, wn);
        dim3 g(cdiv(4096u, 256), cdiv((unsigned)M, 64));
        k_gemm_wmma<<<g, 256, 0, stream>>>(FC1, WB, fc2B, FC2, M, 4096, 4096, 1, 0, 4096);
    }
    {
        size_t wn = (size_t)4096 * 84;
        k_f2bf<<<cdiv((unsigned)wn, 256), 256, 0, stream>>>(clW, WB, wn);
        dim3 g(1, cdiv((unsigned)M, 64));
        k_gemm_wmma<<<g, 256, 0, stream>>>(FC2, WB, clB, OUT + OUT_CLS, M, 84, 4096, 0, 1, 84);
    }
    {
        size_t wn = (size_t)4096 * 21;
        k_f2bf<<<cdiv((unsigned)wn, 256), 256, 0, stream>>>(sfW, WB, wn);
        dim3 g(1, cdiv((unsigned)M, 64));
        k_gemm_wmma<<<g, 256, 0, stream>>>(FC2, WB, sfB, OUT + OUT_SC, M, 21, 4096, 0, 1, 21);
    }
}
